// SOMEBlock_3779571220871
// MI455X (gfx1250) — matrix operand order repaired and checked by reading, not compiled, not run on hardware
//
#include <hip/hip_runtime.h>
#include <hip/hip_bf16.h>

typedef __attribute__((ext_vector_type(16))) __bf16 v16bf;
typedef __attribute__((ext_vector_type(8)))  float  v8f;

#define N_TOK 8192
#define DIM   1024
#define FFN   4096
#define NEXP  8
#define TAU   0.05f
#define BN    16

__device__ __forceinline__ unsigned short f2bf(float x) {
  unsigned u = __builtin_bit_cast(unsigned, x);
  u += 0x7FFFu + ((u >> 16) & 1u);           // round to nearest even
  return (unsigned short)(u >> 16);
}
__device__ __forceinline__ unsigned packbf(float lo, float hi) {
  return (unsigned)f2bf(lo) | ((unsigned)f2bf(hi) << 16);
}
__device__ __forceinline__ v8f v8f_zero() {
  v8f z;
#pragma unroll
  for (int i = 0; i < 8; ++i) z[i] = 0.0f;
  return z;
}
// CDNA5 has a hardware tanh transcendental (V_TANH_F32).
__device__ __forceinline__ float fast_tanh(float z) {
#if __has_builtin(__builtin_amdgcn_tanhf)
  return __builtin_amdgcn_tanhf(z);
#else
  float a = __builtin_fabsf(z);
  float e = __expf(-2.0f * a);                       // v_exp_f32, branch-free
  float r = (1.0f - e) * __builtin_amdgcn_rcpf(1.0f + e);
  return __builtin_copysignf(r, z);
#endif
}

union AFrag { uint4 q[2]; v16bf v; };
union BFrag { uint4 q[2]; v16bf v; };

// ---------------- prep kernels ----------------
__global__ void cvt_h_kernel(const float* __restrict__ src, unsigned* __restrict__ dst, int ndw) {
  int i = blockIdx.x * blockDim.x + threadIdx.x;
  if (i >= ndw) return;
  float2 f = ((const float2*)src)[i];
  dst[i] = packbf(f.x, f.y);
}

// Pack fp32 weights [E][Kdim][Cdim] into WMMA-B-fragment-friendly bf16 layout:
//   dst dword[ ((e*(Kdim/32)+kgrp)*Cdim + c)*16 + idx16 ] =
//       pack( W[e][kgrp*32 + 2*idx16][c], W[e][kgrp*32 + 2*idx16+1][c] )
// so a lane's 8 B-fragment dwords (j=0..7 at k = 16*half + 2j) are contiguous.
__global__ void pack_w_kernel(const float* __restrict__ src, unsigned* __restrict__ dst,
                              int Kdim, int Cdim) {
  size_t gid = (size_t)blockIdx.x * blockDim.x + threadIdx.x;  // = (e*(Kdim/32)+kgrp)*Cdim + c
  size_t perE = (size_t)(Kdim / 32) * Cdim;
  size_t e = gid / perE, rem = gid % perE;
  size_t kgrp = rem / Cdim, c = rem % Cdim;
  const float* s = src + ((size_t)e * Kdim + kgrp * 32) * Cdim + c;

  float v[32];
#pragma unroll
  for (int kk = 0; kk < 32; ++kk) v[kk] = s[(size_t)kk * Cdim];  // coalesced across c

  unsigned dw[16];
#pragma unroll
  for (int i = 0; i < 16; ++i) dw[i] = packbf(v[2 * i], v[2 * i + 1]);

  uint4* d = (uint4*)(dst + gid * 16);   // 64B contiguous per thread -> coalesced
#pragma unroll
  for (int q = 0; q < 4; ++q)
    d[q] = make_uint4(dw[4 * q], dw[4 * q + 1], dw[4 * q + 2], dw[4 * q + 3]);
}

// ---------------- router: softmax(h@Wr+br) with tau mask ----------------
__global__ void router_kernel(const float* __restrict__ h, const float* __restrict__ Wr,
                              const float* __restrict__ br, float* __restrict__ wout) {
  int wave = threadIdx.x >> 5, lane = threadIdx.x & 31;
  int row = blockIdx.x * 8 + wave;
  float acc[NEXP];
#pragma unroll
  for (int e = 0; e < NEXP; ++e) acc[e] = 0.0f;
  const float* hrow = h + (size_t)row * DIM;
  for (int d = lane; d < DIM; d += 32) {
    float hv = hrow[d];
    const float* wr = Wr + (size_t)d * NEXP;
#pragma unroll
    for (int e = 0; e < NEXP; ++e) acc[e] += hv * wr[e];
  }
#pragma unroll
  for (int e = 0; e < NEXP; ++e)
    for (int off = 16; off > 0; off >>= 1)
      acc[e] += __shfl_xor(acc[e], off, 32);
  if (lane == 0) {
    float m = -1e30f;
#pragma unroll
    for (int e = 0; e < NEXP; ++e) { acc[e] += br[e]; m = fmaxf(m, acc[e]); }
    float s = 0.0f;
#pragma unroll
    for (int e = 0; e < NEXP; ++e) { acc[e] = __expf(acc[e] - m); s += acc[e]; }
    float inv = 1.0f / s;
#pragma unroll
    for (int e = 0; e < NEXP; ++e) {
      float w = acc[e] * inv;
      wout[(size_t)row * NEXP + e] = (w > TAU) ? w : 0.0f;
    }
  }
}

// ---------------- fused MoE: per 16-row tile, loop experts ----------------
__global__ __launch_bounds__(256) void moe_kernel(
    const unsigned* __restrict__ hbf,   // [N][DIM/2] bf16 pairs
    const unsigned* __restrict__ w1t,   // packed W1, B-frag layout
    const unsigned* __restrict__ w2t,   // packed W2, B-frag layout
    const float* __restrict__ b1,       // [E][FFN]
    const float* __restrict__ b2,       // [E][DIM]
    const float* __restrict__ rw,       // [N][E] masked router weights
    float* __restrict__ out) {          // [N][DIM]
  constexpr int HROW = DIM / 2 + 4;   // dword row stride (stride%64banks=4 -> conflict-free b128)
  constexpr int QROW = FFN / 2 + 4;
  __shared__ unsigned sh_h[BN * HROW];     // ~33 KB
  __shared__ unsigned sh_hid[BN * QROW];   // ~128 KB (CDNA5: 320KB/WGP)
  __shared__ float sh_w[BN][NEXP];
  __shared__ int sh_active;

  const int tid = threadIdx.x;
  const int wave = tid >> 5, lane = tid & 31;
  const int m = lane & 15, half = lane >> 4;
  const int row0 = blockIdx.x * BN;

  for (int i = tid; i < BN * (DIM / 2); i += 256) {
    int r = i / (DIM / 2), c = i % (DIM / 2);
    sh_h[r * HROW + c] = hbf[(size_t)(row0 + r) * (DIM / 2) + c];
  }
  if (tid < BN * NEXP) {
    int r = tid / NEXP, e = tid % NEXP;
    sh_w[r][e] = rw[(size_t)(row0 + r) * NEXP + e];
  }
  __syncthreads();

  v8f acc[8];
#pragma unroll
  for (int t = 0; t < 8; ++t) acc[t] = v8f_zero();

  for (int e = 0; e < NEXP; ++e) {
    if (tid == 0) {
      int a = 0;
      for (int r = 0; r < BN; ++r) a |= (sh_w[r][e] != 0.0f);
      sh_active = a;
    }
    __syncthreads();
    int act = sh_active;
    __syncthreads();
    if (!act) continue;   // uniform: whole expert inactive for this tile

    const unsigned* w1e = w1t + (size_t)e * (DIM / 32) * FFN * 16;
    const unsigned* w2e = w2t + (size_t)e * (FFN / 32) * DIM * 16;
    const float* b1e = b1 + (size_t)e * FFN;

    // ---- layer 1: sh_hid = f2bf( w[r] * gelu(h @ W1e + b1e) ) ----
    for (int c = 0; c < 8; ++c) {
      const int f_base = (wave + 8 * c) * 64;    // 4 ftiles of 16 cols
      v8f a1[4];
#pragma unroll
      for (int t = 0; t < 4; ++t) a1[t] = v8f_zero();
#pragma unroll 1
      for (int k0 = 0; k0 < DIM; k0 += 32) {
        AFrag A;
        const unsigned* ha = &sh_h[m * HROW + (k0 >> 1) + half * 8];
        A.q[0] = *(const uint4*)(ha);
        A.q[1] = *(const uint4*)(ha + 4);
        const size_t kbase = (size_t)(k0 >> 5) * FFN;
#pragma unroll
        for (int t = 0; t < 4; ++t) {
          BFrag B;
          const uint4* wb = (const uint4*)(w1e + (kbase + (f_base + t * 16 + m)) * 16 + half * 8);
          B.q[0] = wb[0];
          B.q[1] = wb[1];
          a1[t] = __builtin_amdgcn_wmma_f32_16x16x32_bf16(false, A.v, false, B.v,
                                                          (short)0, a1[t], false, false);
        }
      }
#pragma unroll
      for (int t = 0; t < 4; ++t) {
        const int col = f_base + t * 16 + m;
        const float bias = b1e[col];
#pragma unroll
        for (int i = 0; i < 8; ++i) {
          const int r = i + half * 8;
          float x = a1[t][i] + bias;
          float g = 0.5f * x * (1.0f + fast_tanh(0.7978845608f * (x + 0.044715f * x * x * x)));
          ((unsigned short*)sh_hid)[r * (2 * QROW) + col] = f2bf(g * sh_w[r][e]);
        }
      }
    }
    __syncthreads();

    // ---- layer 2: acc += (scaled hid) @ W2e ----
#pragma unroll 1
    for (int k0 = 0; k0 < FFN; k0 += 32) {
      AFrag A;
      const unsigned* ha = &sh_hid[m * QROW + (k0 >> 1) + half * 8];
      A.q[0] = *(const uint4*)(ha);
      A.q[1] = *(const uint4*)(ha + 4);
      const size_t kbase = (size_t)(k0 >> 5) * DIM;
#pragma unroll
      for (int t = 0; t < 8; ++t) {
        const int d0 = (wave + 8 * t) * 16;
        BFrag B;
        const uint4* wb = (const uint4*)(w2e + (kbase + (d0 + m)) * 16 + half * 8);
        B.q[0] = wb[0];
        B.q[1] = wb[1];
        acc[t] = __builtin_amdgcn_wmma_f32_16x16x32_bf16(false, A.v, false, B.v,
                                                         (short)0, acc[t], false, false);
      }
    }
    __syncthreads();
  }

  // ---- store: out = acc + sum_e w[r][e] * b2[e][d] ----
#pragma unroll
  for (int t = 0; t < 8; ++t) {
    const int d0 = (wave + 8 * t) * 16 + m;
#pragma unroll
    for (int i = 0; i < 8; ++i) {
      const int r = i + half * 8;
      float bias = 0.0f;
#pragma unroll
      for (int e = 0; e < NEXP; ++e) bias += sh_w[r][e] * b2[(size_t)e * DIM + d0];
      out[(size_t)(row0 + r) * DIM + d0] = acc[t][i] + bias;
    }
  }
}

extern "C" void kernel_launch(void* const* d_in, const int* in_sizes, int n_in,
                              void* d_out, int out_size, void* d_ws, size_t ws_size,
                              hipStream_t stream) {
  const float* h  = (const float*)d_in[0];
  const float* Wr = (const float*)d_in[1];
  const float* br = (const float*)d_in[2];
  const float* W1 = (const float*)d_in[3];
  const float* b1 = (const float*)d_in[4];
  const float* W2 = (const float*)d_in[5];
  const float* b2 = (const float*)d_in[6];
  float* out = (float*)d_out;

  char* ws = (char*)d_ws;
  unsigned* hbf = (unsigned*)ws;
  size_t off = (size_t)N_TOK * (DIM / 2) * 4;
  unsigned* w1t = (unsigned*)(ws + off); off += (size_t)NEXP * (DIM / 2) * FFN * 4;
  unsigned* w2t = (unsigned*)(ws + off); off += (size_t)NEXP * (FFN / 2) * DIM * 4;
  float* rw = (float*)(ws + off);

  cvt_h_kernel<<<(N_TOK * DIM / 2) / 256, 256, 0, stream>>>(h, hbf, N_TOK * DIM / 2);
  // one thread per (e, kgrp, column): W1: 8*32*4096 = 1M, W2: 8*128*1024 = 1M
  pack_w_kernel<<<(int)(((size_t)NEXP * (DIM / 32) * FFN) / 256), 256, 0, stream>>>(W1, w1t, DIM, FFN);
  pack_w_kernel<<<(int)(((size_t)NEXP * (FFN / 32) * DIM) / 256), 256, 0, stream>>>(W2, w2t, FFN, DIM);
  router_kernel<<<N_TOK / 8, 256, 0, stream>>>(h, Wr, br, rw);
  moe_kernel<<<N_TOK / BN, 256, 0, stream>>>(hbf, w1t, w2t, b1, b2, rw, out);
}
